// RelativePositionMultiHeadAttention_50027779064526
// MI455X (gfx1250) — matrix operand order repaired and checked by reading, not compiled, not run on hardware
//
#include <hip/hip_runtime.h>
#include <cstdint>

// ---------------------------------------------------------------------------
// Types for CDNA5 WMMA (wave32, v_wmma_f32_16x16x32_f16) and TDM descriptors
// ---------------------------------------------------------------------------
typedef _Float16 h16;
typedef __attribute__((ext_vector_type(8)))  _Float16 h8v;
typedef __attribute__((ext_vector_type(16))) _Float16 h16v;
typedef __attribute__((ext_vector_type(8)))  float    f8v;
typedef __attribute__((ext_vector_type(4)))  unsigned int u32x4;
typedef __attribute__((ext_vector_type(8)))  int i32x8;
typedef __attribute__((ext_vector_type(4)))  int i32x4;

union V16 { h16v v; h8v h[2]; };

static __device__ __forceinline__ f8v wmma16(h16v a, h16v b, f8v c) {
  return __builtin_amdgcn_wmma_f32_16x16x32_f16(false, a, false, b, (short)0, c,
                                                false, false);
}

#define BATCH 8
#define CCH   512
#define T_SEQ 1024
#define HEADS 8
#define DKH   64
#define RWIN  4

// ---------------------------------------------------------------------------
// TDM: 2D tile load, data_size=8B. Tile = tile_rows x (tile0_units*8B),
// tensor row stride = stride_units*8B. Descriptors built per CDNA5 ISA 8.3/8.4.
// ---------------------------------------------------------------------------
static __device__ __forceinline__ void tdm_load_2d(const void* gsrc, void* ldsdst,
                                                   unsigned dim0_units,
                                                   unsigned rows_total,
                                                   unsigned tile0_units,
                                                   unsigned tile_rows,
                                                   unsigned stride_units) {
  uint64_t ga = (uint64_t)(uintptr_t)gsrc;
  uint32_t la = (uint32_t)(uintptr_t)ldsdst;   // low 32 bits = LDS byte address
  u32x4 g0;
  g0.x = 1u;                                   // count=1, user mode, no gather
  g0.y = la;                                   // lds_addr
  g0.z = (uint32_t)ga;                         // global_addr[31:0]
  g0.w = (uint32_t)((ga >> 32) & 0x1FFFFFFu) | (2u << 30);  // addr[56:32] | type=2
  i32x8 g1;
  g1[0] = (int)(3u << 16);                     // wg_mask=0, data_size=8B, no flags
  g1[1] = (int)((dim0_units & 0xFFFFu) << 16); // tensor_dim0[15:0] @ bits[63:48]
  g1[2] = (int)(((dim0_units >> 16) & 0xFFFFu) | ((rows_total & 0xFFFFu) << 16));
  g1[3] = (int)(((rows_total >> 16) & 0xFFFFu) | ((tile0_units & 0xFFFFu) << 16));
  g1[4] = (int)(tile_rows & 0xFFFFu);          // tile_dim1; tile_dim2=0 (2D)
  g1[5] = (int)stride_units;                   // tensor_dim0_stride[31:0]
  g1[6] = 0;
  g1[7] = 0;
  i32x4 gz = {0, 0, 0, 0};                     // groups 2/3: unused dims = 0
#if __has_builtin(__builtin_amdgcn_tensor_load_to_lds)
#if defined(__clang_major__) && __clang_major__ >= 23
  i32x8 gz8 = {0, 0, 0, 0, 0, 0, 0, 0};
  __builtin_amdgcn_tensor_load_to_lds(g0, g1, gz, gz, gz8, 0);
#else
  __builtin_amdgcn_tensor_load_to_lds(g0, g1, gz, gz, 0);
#endif
#else
  asm volatile("tensor_load_to_lds %0, %1" :: "s"(g0), "s"(g1) : "memory");
#endif
}

static __device__ __forceinline__ void wait_tensorcnt1() {
#if __has_builtin(__builtin_amdgcn_s_wait_tensorcnt)
  __builtin_amdgcn_s_wait_tensorcnt(1);
#else
  asm volatile("s_wait_tensorcnt 0x1" ::: "memory");
#endif
}
static __device__ __forceinline__ void wait_tensorcnt0() {
#if __has_builtin(__builtin_amdgcn_s_wait_tensorcnt)
  __builtin_amdgcn_s_wait_tensorcnt(0);
#else
  asm volatile("s_wait_tensorcnt 0x0" ::: "memory");
#endif
}

// ---------------------------------------------------------------------------
// Prepass: f32 [B][C][T] -> f16 [B][T][C]  (transpose for contiguous K-dim)
// ---------------------------------------------------------------------------
__global__ void k_transpose_cvt(const float* __restrict__ in, h16* __restrict__ out) {
  size_t idx = (size_t)blockIdx.x * 256 + threadIdx.x;
  size_t total = (size_t)BATCH * CCH * T_SEQ;
  if (idx >= total) return;
  int c = (int)(idx % CCH);
  size_t r = idx / CCH;
  int t = (int)(r % T_SEQ);
  int b = (int)(r / T_SEQ);
  out[idx] = (h16)in[((size_t)b * CCH + c) * T_SEQ + t];
}

__global__ void k_cvt_f32_f16(const float* __restrict__ in, h16* __restrict__ out, int n) {
  int i = blockIdx.x * 256 + threadIdx.x;
  if (i < n) out[i] = (h16)in[i];
}

// ---------------------------------------------------------------------------
// QKV projection: D[t,o] = sum_c xT[b][t][c] * W[o][c]; (+bias)*scale
// Wave -> 16 t-rows x 64 o-cols; K-loop software-pipelined in registers.
// Output layout: [B][H][T][DK] f16 (q pre-scaled by 1/sqrt(DK)).
// ---------------------------------------------------------------------------
__global__ __launch_bounds__(128) void k_qkv_gemm(const h16* __restrict__ xT,
                                                  const h16* __restrict__ W,
                                                  const float* __restrict__ bias,
                                                  h16* __restrict__ out, float scale) {
  int tid = threadIdx.x, wave = tid >> 5, lane = tid & 31;
  int l16 = lane & 15, hs = lane >> 4;
  int wid = blockIdx.x * 4 + wave;          // 4096 waves total
  int og = wid & 7;
  int tt = (wid >> 3) & 63;
  int b  = wid >> 9;
  int t0 = tt * 16, o0 = og * 64;

  f8v zero = {};
  f8v acc[4];
  for (int j = 0; j < 4; ++j) acc[j] = zero;

  const h16* arow = xT + ((size_t)b * T_SEQ + t0 + l16) * CCH;
  const h16* brow0 = W + (size_t)(o0 + l16) * CCH + 16 * hs;

  V16 a_cur, b_cur[4], a_nxt, b_nxt[4];
  a_cur.h[0] = *(const h8v*)(arow + 16 * hs);
  a_cur.h[1] = *(const h8v*)(arow + 16 * hs + 8);
  for (int j = 0; j < 4; ++j) {
    const h16* p = brow0 + (size_t)(j * 16) * CCH;
    b_cur[j].h[0] = *(const h8v*)p;
    b_cur[j].h[1] = *(const h8v*)(p + 8);
  }
  for (int cc = 0; cc < 16; ++cc) {
    if (cc < 15) {                           // prefetch next K-chunk into regs
      int c1 = (cc + 1) * 32;
      a_nxt.h[0] = *(const h8v*)(arow + c1 + 16 * hs);
      a_nxt.h[1] = *(const h8v*)(arow + c1 + 16 * hs + 8);
      for (int j = 0; j < 4; ++j) {
        const h16* p = brow0 + (size_t)(j * 16) * CCH + c1;
        b_nxt[j].h[0] = *(const h8v*)p;
        b_nxt[j].h[1] = *(const h8v*)(p + 8);
      }
    }
    for (int j = 0; j < 4; ++j) acc[j] = wmma16(a_cur.v, b_cur[j].v, acc[j]);
    a_cur = a_nxt;
    for (int j = 0; j < 4; ++j) b_cur[j] = b_nxt[j];
  }
  for (int j = 0; j < 4; ++j) {
    int o = o0 + j * 16 + l16;
    float bv = bias[o];
    int hh = o >> 6, dk = o & 63;
    h16* obase = out + ((size_t)(b * HEADS + hh)) * T_SEQ * DKH + dk;
    for (int i = 0; i < 8; ++i) {
      int t = t0 + i + 8 * hs;
      obase[(size_t)t * DKH] = (h16)((acc[j][i] + bv) * scale);
    }
  }
}

// ---------------------------------------------------------------------------
// Fused flash attention. 4 waves/block, 16 q-rows each (64 rows/block).
// K tiles:   TDM tensor_load_to_lds, double-buffered, 2 tiles ahead.
// V tiles:   cooperative load, stored TRANSPOSED in LDS (padded rows) so the
//            PV B-operand gather is two ds_load_b128; double-buffered.
// Per tile: QK^T WMMA + rel-k band + bias table + mask, online softmax,
//           P via LDS, PV WMMA, rel-v band.
// ---------------------------------------------------------------------------
__global__ __launch_bounds__(128) void k_flash(const h16* __restrict__ q,
                                               const h16* __restrict__ kM,
                                               const h16* __restrict__ vM,
                                               const int* __restrict__ mask,
                                               const float* __restrict__ relk,
                                               const float* __restrict__ relv,
                                               h16* __restrict__ Z) {
  __shared__ alignas(16) h16 Kt[2][32][64];    // key-major, TDM destination
  __shared__ alignas(16) h16 VtT[2][64][40];   // dk-major (transposed), padded
  __shared__ alignas(16) h16 Pt[4][16][32];
  __shared__ alignas(16) h16 Qt[4][16][64];
  __shared__ float btab[T_SEQ];

  int tid = threadIdx.x, wave = tid >> 5, lane = tid & 31;
  int l16 = lane & 15, hs = lane >> 4;

  int blk = blockIdx.x;                        // B*H*(T/64) = 1024 blocks
  int qblk = blk & 15;
  int h = (blk >> 4) & 7;
  int b = blk >> 7;
  int qt0 = qblk * 64 + wave * 16;

  const h16* qbase = q  + ((size_t)(b * HEADS + h)) * T_SEQ * DKH;
  const h16* kbase = kM + ((size_t)(b * HEADS + h)) * T_SEQ * DKH;
  const h16* vbase = vM + ((size_t)(b * HEADS + h)) * T_SEQ * DKH;
  const int* mbase = mask + (size_t)b * T_SEQ * T_SEQ;

  // proximity-bias lookup table: -log1p(|t-s|)
  for (int i = tid; i < T_SEQ; i += 128) btab[i] = -__logf(1.0f + (float)i);

  // stash this wave's Q tile (for the rel-k band dot products)
  for (int i = lane; i < 16 * 8; i += 32) {
    int r = i >> 3, g = i & 7;
    *(h8v*)&Qt[wave][r][g * 8] = *(const h8v*)(qbase + (size_t)(qt0 + r) * DKH + g * 8);
  }

  // Q as A operand (two K=32 chunks over DK=64)
  V16 qa[2];
  {
    const h16* qr = qbase + (size_t)(qt0 + l16) * DKH;
    for (int c = 0; c < 2; ++c) {
      qa[c].h[0] = *(const h8v*)(qr + c * 32 + 16 * hs);
      qa[c].h[1] = *(const h8v*)(qr + c * 32 + 16 * hs + 8);
    }
  }

  // stage V tiles 0 and 1 (transposed), issue TDM for K tiles 0 and 1
  for (int tile = 0; tile < 2; ++tile) {
    for (int i = tid; i < 32 * 8; i += 128) {
      int r = i >> 3, g = i & 7;
      h8v vv = *(const h8v*)(vbase + (size_t)(tile * 32 + r) * DKH + g * 8);
#pragma unroll
      for (int e = 0; e < 8; ++e) VtT[tile][g * 8 + e][r] = vv[e];
    }
  }
  if (wave == 0) {
    tdm_load_2d(kbase, &Kt[0][0][0], 16, T_SEQ, 16, 32, 16);
    tdm_load_2d(kbase + 32 * DKH, &Kt[1][0][0], 16, T_SEQ, 16, 32, 16);
  }

  f8v zero = {};
  f8v acc[4];
  float mrow[8], lrow[8];
  for (int j = 0; j < 4; ++j) acc[j] = zero;
  for (int i = 0; i < 8; ++i) { mrow[i] = -3.0e38f; lrow[i] = 0.0f; }

  for (int it = 0; it < T_SEQ / 32; ++it) {
    int kt = it * 32, buf = it & 1;
    if (wave == 0) {
      if (it < 30) wait_tensorcnt1();          // K tile `it` has landed
      else         wait_tensorcnt0();
    }
    __syncthreads();                           // K TDM + V stores visible to all

    // ---- scores: two 16x16 C-tiles ----
    f8v s[2];
    for (int n2 = 0; n2 < 2; ++n2) {
      s[n2] = zero;
      for (int c = 0; c < 2; ++c) {
        const h16* kr = &Kt[buf][n2 * 16 + l16][c * 32 + 16 * hs];
        V16 kb;
        kb.h[0] = *(const h8v*)kr;
        kb.h[1] = *(const h8v*)(kr + 8);
        s[n2] = wmma16(qa[c].v, kb.v, s[n2]);
      }
    }

    bool band = (kt <= qt0 + 15 + RWIN) && (kt + 31 >= qt0 - RWIN);

    // ---- rel-k band + proximity bias + mask ----
    for (int n2 = 0; n2 < 2; ++n2) {
      int scol = kt + n2 * 16 + l16;
      for (int i = 0; i < 8; ++i) {
        int trow = qt0 + i + 8 * hs;
        float sv = s[n2][i];
        int d = scol - trow;
        if (band && d >= -RWIN && d <= RWIN) {
          const float* rk = relk + (size_t)(d + RWIN) * DKH;
          float dot = 0.0f;
          for (int kk = 0; kk < DKH; ++kk)
            dot += (float)Qt[wave][trow - qt0][kk] * rk[kk];  // q pre-scaled
          sv += dot;
        }
        int ad = d < 0 ? -d : d;
        sv += btab[ad];
        if (mbase[(size_t)trow * T_SEQ + scol] == 0) sv = -10000.0f;
        s[n2][i] = sv;
      }
    }

    // ---- online softmax (row reductions across 16-lane halves) ----
    for (int i = 0; i < 8; ++i) {
      float tmax = fmaxf(s[0][i], s[1][i]);
      for (int off = 8; off >= 1; off >>= 1)
        tmax = fmaxf(tmax, __shfl_xor(tmax, off, 32));
      float mnew = fmaxf(mrow[i], tmax);
      float corr = __expf(mrow[i] - mnew);
      float p0 = __expf(s[0][i] - mnew);
      float p1 = __expf(s[1][i] - mnew);
      float rs = p0 + p1;
      for (int off = 8; off >= 1; off >>= 1)
        rs += __shfl_xor(rs, off, 32);
      lrow[i] = lrow[i] * corr + rs;
      mrow[i] = mnew;
      for (int j = 0; j < 4; ++j) acc[j][i] *= corr;
      Pt[wave][i + 8 * hs][l16]      = (h16)p0;  // C-layout -> row-major LDS
      Pt[wave][i + 8 * hs][16 + l16] = (h16)p1;
    }
    __syncthreads();                           // P visible (and orders LDS)

    // ---- PV: acc(16x64) += P(16x32) @ V(32x64) ----
    V16 pa;
    pa.h[0] = *(const h8v*)&Pt[wave][l16][16 * hs];
    pa.h[1] = *(const h8v*)&Pt[wave][l16][16 * hs + 8];
    for (int j = 0; j < 4; ++j) {
      const h16* vr = &VtT[buf][j * 16 + l16][16 * hs];
      V16 vb;
      vb.h[0] = *(const h8v*)vr;               // B: K = 16*hs + 0..7
      vb.h[1] = *(const h8v*)(vr + 8);         //    K = 16*hs + 8..15
      acc[j] = wmma16(pa.v, vb.v, acc[j]);
    }

    // ---- rel-v band: acc[t] += sum_{|s-t|<=4} P[t,s] * emb_rel_v[s-t+4] ----
    if (band) {
      for (int j = 0; j < 4; ++j) {
        int dcol = j * 16 + l16;
        for (int i = 0; i < 8; ++i) {
          int trow = qt0 + i + 8 * hs;
          int slo = trow - RWIN < kt ? kt : trow - RWIN;
          int shi = trow + RWIN > kt + 31 ? kt + 31 : trow + RWIN;
          float a = 0.0f;
          for (int ss = slo; ss <= shi; ++ss)
            a += (float)Pt[wave][i + 8 * hs][ss - kt] *
                 relv[(size_t)(ss - trow + RWIN) * DKH + dcol];
          acc[j][i] += a;
        }
      }
    }
    __syncthreads();                           // all reads of buf done

    // ---- software pipeline: stage tile it+2 into the freed buffer ----
    if (it + 2 < T_SEQ / 32) {
      int ktn = kt + 64;
      for (int i = tid; i < 32 * 8; i += 128) {
        int r = i >> 3, g = i & 7;
        h8v vv = *(const h8v*)(vbase + (size_t)(ktn + r) * DKH + g * 8);
#pragma unroll
        for (int e = 0; e < 8; ++e) VtT[buf][g * 8 + e][r] = vv[e];
      }
      if (wave == 0)
        tdm_load_2d(kbase + (size_t)ktn * DKH, &Kt[buf][0][0], 16, T_SEQ, 16, 32, 16);
    }
  }

  // epilogue: normalize and store Z[b][t][h*64 + d]
  for (int j = 0; j < 4; ++j) {
    int cidx = h * 64 + j * 16 + l16;
    for (int i = 0; i < 8; ++i) {
      int trow = qt0 + i + 8 * hs;
      float o = acc[j][i] / lrow[i];
      Z[((size_t)b * T_SEQ + trow) * CCH + cidx] = (h16)o;
    }
  }
}

// ---------------------------------------------------------------------------
// Output projection: out[b][o][t] = sum_c Wo[o][c] * Z[b][t][c] + bo[o]  (f32)
// ---------------------------------------------------------------------------
__global__ __launch_bounds__(128) void k_proj(const h16* __restrict__ Z,
                                              const h16* __restrict__ Wo,
                                              const float* __restrict__ bo,
                                              float* __restrict__ out) {
  int tid = threadIdx.x, wave = tid >> 5, lane = tid & 31;
  int l16 = lane & 15, hs = lane >> 4;
  int wid = blockIdx.x * 4 + wave;
  int tg = wid & 15;
  int ot = (wid >> 4) & 31;
  int b  = wid >> 9;
  int o0 = ot * 16, t0 = tg * 64;

  f8v zero = {};
  f8v acc[4];
  for (int j = 0; j < 4; ++j) acc[j] = zero;

  const h16* arow = Wo + (size_t)(o0 + l16) * CCH;
  const h16* brow0 = Z + ((size_t)b * T_SEQ + t0 + l16) * CCH + 16 * hs;

  V16 a_cur, b_cur[4], a_nxt, b_nxt[4];
  a_cur.h[0] = *(const h8v*)(arow + 16 * hs);
  a_cur.h[1] = *(const h8v*)(arow + 16 * hs + 8);
  for (int j = 0; j < 4; ++j) {
    const h16* p = brow0 + (size_t)(j * 16) * CCH;
    b_cur[j].h[0] = *(const h8v*)p;
    b_cur[j].h[1] = *(const h8v*)(p + 8);
  }
  for (int cc = 0; cc < 16; ++cc) {
    if (cc < 15) {
      int c1 = (cc + 1) * 32;
      a_nxt.h[0] = *(const h8v*)(arow + c1 + 16 * hs);
      a_nxt.h[1] = *(const h8v*)(arow + c1 + 16 * hs + 8);
      for (int j = 0; j < 4; ++j) {
        const h16* p = brow0 + (size_t)(j * 16) * CCH + c1;
        b_nxt[j].h[0] = *(const h8v*)p;
        b_nxt[j].h[1] = *(const h8v*)(p + 8);
      }
    }
    for (int j = 0; j < 4; ++j) acc[j] = wmma16(a_cur.v, b_cur[j].v, acc[j]);
    a_cur = a_nxt;
    for (int j = 0; j < 4; ++j) b_cur[j] = b_nxt[j];
  }
  for (int i = 0; i < 8; ++i) {
    int o = o0 + i + 8 * hs;
    float bv = bo[o];
    for (int j = 0; j < 4; ++j) {
      int t = t0 + j * 16 + l16;
      out[((size_t)b * CCH + o) * T_SEQ + t] = acc[j][i] + bv;
    }
  }
}

// ---------------------------------------------------------------------------
// Launch
// ---------------------------------------------------------------------------
extern "C" void kernel_launch(void* const* d_in, const int* in_sizes, int n_in,
                              void* d_out, int out_size, void* d_ws, size_t ws_size,
                              hipStream_t stream) {
  const float* x    = (const float*)d_in[0];
  const float* c    = (const float*)d_in[1];
  const int*   msk  = (const int*)d_in[2];
  const float* Wq   = (const float*)d_in[3];
  const float* bq   = (const float*)d_in[4];
  const float* Wk   = (const float*)d_in[5];
  const float* bk   = (const float*)d_in[6];
  const float* Wv   = (const float*)d_in[7];
  const float* bv   = (const float*)d_in[8];
  const float* Wo   = (const float*)d_in[9];
  const float* bo   = (const float*)d_in[10];
  const float* relk = (const float*)d_in[11];
  const float* relv = (const float*)d_in[12];
  float* out = (float*)d_out;

  char* ws = (char*)d_ws;
  size_t off = 0;
  auto carve = [&](size_t bytes) -> void* {
    void* p = ws + off;
    off += (bytes + 255) & ~(size_t)255;
    return p;
  };
  const size_t ACT = (size_t)BATCH * T_SEQ * CCH;
  const size_t WEL = (size_t)CCH * CCH;
  h16* xT  = (h16*)carve(ACT * 2);
  h16* cT  = (h16*)carve(ACT * 2);
  h16* WqH = (h16*)carve(WEL * 2);
  h16* WkH = (h16*)carve(WEL * 2);
  h16* WvH = (h16*)carve(WEL * 2);
  h16* WoH = (h16*)carve(WEL * 2);
  h16* qws = (h16*)carve(ACT * 2);
  h16* kws = (h16*)carve(ACT * 2);
  h16* vws = (h16*)carve(ACT * 2);
  h16* Zws = (h16*)carve(ACT * 2);

  int tb = (int)((ACT + 255) / 256);
  k_transpose_cvt<<<tb, 256, 0, stream>>>(x, xT);
  k_transpose_cvt<<<tb, 256, 0, stream>>>(c, cT);
  int wb = (int)((WEL + 255) / 256);
  k_cvt_f32_f16<<<wb, 256, 0, stream>>>(Wq, WqH, (int)WEL);
  k_cvt_f32_f16<<<wb, 256, 0, stream>>>(Wk, WkH, (int)WEL);
  k_cvt_f32_f16<<<wb, 256, 0, stream>>>(Wv, WvH, (int)WEL);
  k_cvt_f32_f16<<<wb, 256, 0, stream>>>(Wo, WoH, (int)WEL);

  k_qkv_gemm<<<1024, 128, 0, stream>>>(xT, WqH, bq, qws, 0.125f);
  k_qkv_gemm<<<1024, 128, 0, stream>>>(cT, WkH, bk, kws, 1.0f);
  k_qkv_gemm<<<1024, 128, 0, stream>>>(cT, WvH, bv, vws, 1.0f);

  k_flash<<<1024, 128, 0, stream>>>(qws, kws, vws, msk, relk, relv, Zws);

  k_proj<<<1024, 128, 0, stream>>>(Zws, WoH, bo, out);
}
